// CustomConv_58050777973327
// MI455X (gfx1250) — matrix operand order repaired and checked by reading, not compiled, not run on hardware
//
#include <hip/hip_runtime.h>

typedef __bf16 bf16;
typedef __attribute__((ext_vector_type(16))) __bf16 v16bf;
typedef __attribute__((ext_vector_type(8)))  __bf16 v8bf;
typedef __attribute__((ext_vector_type(8)))  float  v8f;

#define CI 128
#define CO 256
#define HH 64
#define WW 64
#define NB 32

// Staged x tile in LDS: 3 halo rows x 66 halo cols x 32 channels.
#define XROWS 3
#define XCOLS 66
#define XCH   32
#define XTILE (XROWS * XCOLS * XCH)   // 6336 elements per plane

// Prologue: split fp32 weights into bf16 hi/lo planes, transposed to
// Wt[kpos][co][ci] so the WMMA A-fragment reads are contiguous in ci (K).
__global__ __launch_bounds__(256) void wt_transform(const float* __restrict__ wsrc,
                                                    bf16* __restrict__ whi,
                                                    bf16* __restrict__ wlo) {
  const int idx = blockIdx.x * 256 + threadIdx.x;  // 9*256*128 = 294912 total
  const int ci   = idx & 127;
  const int co   = (idx >> 7) & 255;
  const int kpos = idx >> 15;                       // 0..8
  const float v  = wsrc[(co * CI + ci) * 9 + kpos]; // OIHW
  const bf16 h   = (bf16)v;
  const float rem = v - (float)h;
  const int o = (kpos * CO + co) * CI + ci;
  whi[o] = h;
  wlo[o] = (bf16)rem;
}

// Implicit-GEMM conv: D[co, w] += Wt[kpos][co][ci] * X[ci, w+dw, h+dh]
// One block: (b, h, co-block of 128) -> 128 c_out x 64 pixels.
// x is staged once per ci-block (3 rows x 66 cols halo tile); all 9 kernel
// positions then run out of LDS with shifted row/col offsets.
__global__ __launch_bounds__(256) void conv_wmma(const float* __restrict__ x,
                                                 const bf16* __restrict__ whi,
                                                 const bf16* __restrict__ wlo,
                                                 const float* __restrict__ bias,
                                                 float* __restrict__ out) {
  __shared__ __align__(32) bf16 lXhi[XTILE];  // [hrow][wi][ci], ~12.4 KB
  __shared__ __align__(32) bf16 lXlo[XTILE];

  const int coblk = blockIdx.x;   // 0..1
  const int h     = blockIdx.y;   // 0..63
  const int b     = blockIdx.z;   // 0..31

  const int tid  = threadIdx.x;
  const int lane = tid & 31;
  const int wid  = tid >> 5;      // 8 waves
  const int wave_m = wid & 3;     // co quadrant (32 each)
  const int wave_n = wid >> 2;    // pixel half (32 each)
  const int co_wave = coblk * 128 + wave_m * 32;
  const int w_wave  = wave_n * 32;

  const int  lco    = lane & 15;
  const bool hihalf = lane >= 16;

  v8f acc[2][2] = {};

  for (int cib = 0; cib < 4; ++cib) {
    const int ci0 = cib << 5;

    __syncthreads();  // previous ci-block's LDS fully consumed

    // ---- stage 3x66x32 halo tile, fp32 -> bf16 hi/lo split ----
    for (int t = tid; t < XTILE; t += 256) {
      const int ci   = t & (XCH - 1);
      const int rest = t >> 5;              // hrow*66 + wi
      const int wi   = rest % XCOLS;
      const int hrow = rest / XCOLS;
      const int hin  = h + hrow - 1;
      const int win  = wi - 1;
      float v = 0.0f;
      if ((unsigned)hin < (unsigned)HH && (unsigned)win < (unsigned)WW)
        v = x[(((long)b * CI + ci0 + ci) * HH + hin) * WW + win];
      const bf16 hv = (bf16)v;
      lXhi[t] = hv;
      lXlo[t] = (bf16)(v - (float)hv);
    }

    // warm caches for the next ci-block (global_prefetch_b8)
    if (cib < 3) {
      const long pidx = (((long)b * CI + ci0 + 32 + (tid >> 3)) * HH + h) * WW + (tid & 7) * 8;
      __builtin_prefetch(x + pidx, 0, 1);
    }

    __syncthreads();

    // ---- all 9 kernel positions from the staged tile ----
    for (int kpos = 0; kpos < 9; ++kpos) {
      const int kh = kpos / 3;        // LDS halo row 0..2
      const int dw = kpos % 3 - 1;    // column shift -1..1

      // A fragments (weights, 16x32 bf16) from L2-resident Wt.
      // ISA layout: lanes 0-15 hold K 0..7 & 16..23, lanes 16-31 hold 8..15 & 24..31.
      v16bf aHi[2], aLo[2];
      const bf16* wbase_hi = whi + (long)(kpos * CO) * CI;
      const bf16* wbase_lo = wlo + (long)(kpos * CO) * CI;
      const int kofs = ci0 + (hihalf ? 8 : 0);
#pragma unroll
      for (int mi = 0; mi < 2; ++mi) {
        const int co = co_wave + mi * 16 + lco;
        const v8bf* ph = (const v8bf*)(wbase_hi + co * CI + kofs);
        v8bf a0 = ph[0];
        v8bf a1 = ph[2];  // +16 K elements
        aHi[mi] = __builtin_shufflevector(a0, a1, 0,1,2,3,4,5,6,7,8,9,10,11,12,13,14,15);
        const v8bf* pl = (const v8bf*)(wbase_lo + co * CI + kofs);
        v8bf b0 = pl[0];
        v8bf b1 = pl[2];
        aLo[mi] = __builtin_shufflevector(b0, b1, 0,1,2,3,4,5,6,7,8,9,10,11,12,13,14,15);
      }

      // B fragments (x, 32x16 bf16) from LDS with shifted row/col.
      // ISA layout: lanes 0-15 hold K 0..7 & 16..23, lanes 16-31 hold 8..15 & 24..31.
      v16bf bHi[2], bLo[2];
#pragma unroll
      for (int ni = 0; ni < 2; ++ni) {
        const int wrow = w_wave + ni * 16 + lco + dw + 1;   // halo col 0..65
        const int o = (kh * XCOLS + wrow) * XCH + (hihalf ? 8 : 0);  // 16B aligned
        const v8bf* qh = (const v8bf*)(lXhi + o);
        bHi[ni] = __builtin_shufflevector(qh[0], qh[2], 0,1,2,3,4,5,6,7,8,9,10,11,12,13,14,15);
        const v8bf* ql = (const v8bf*)(lXlo + o);
        bLo[ni] = __builtin_shufflevector(ql[0], ql[2], 0,1,2,3,4,5,6,7,8,9,10,11,12,13,14,15);
      }

      // hi/lo split products: hi*hi + hi*lo + lo*hi (~fp32 accuracy)
#pragma unroll
      for (int mi = 0; mi < 2; ++mi)
#pragma unroll
        for (int ni = 0; ni < 2; ++ni) {
          acc[mi][ni] = __builtin_amdgcn_wmma_f32_16x16x32_bf16(
              false, aHi[mi], false, bHi[ni], (short)0, acc[mi][ni], false, false);
          acc[mi][ni] = __builtin_amdgcn_wmma_f32_16x16x32_bf16(
              false, aHi[mi], false, bLo[ni], (short)0, acc[mi][ni], false, false);
          acc[mi][ni] = __builtin_amdgcn_wmma_f32_16x16x32_bf16(
              false, aLo[mi], false, bHi[ni], (short)0, acc[mi][ni], false, false);
        }
    }
  }

  // ---- store: D VGPR r holds M=r (lanes 0-15) / M=r+8 (lanes 16-31), N=lane%16.
  // N runs along w -> each half-wave writes 64 contiguous bytes.
  const int moff = hihalf ? 8 : 0;
#pragma unroll
  for (int mi = 0; mi < 2; ++mi) {
#pragma unroll
    for (int ni = 0; ni < 2; ++ni) {
      const int wcol = w_wave + ni * 16 + lco;
#pragma unroll
      for (int r = 0; r < 8; ++r) {
        const int co = co_wave + mi * 16 + r + moff;
        out[(((long)b * CO + co) * HH + h) * WW + wcol] = acc[mi][ni][r] + bias[co];
      }
    }
  }
}

extern "C" void kernel_launch(void* const* d_in, const int* in_sizes, int n_in,
                              void* d_out, int out_size, void* d_ws, size_t ws_size,
                              hipStream_t stream) {
  (void)in_sizes; (void)n_in; (void)out_size;
  const float* x    = (const float*)d_in[0];
  const float* wgt  = (const float*)d_in[1];
  const float* bias = (const float*)d_in[2];
  float* out = (float*)d_out;

  const size_t wt_elems = (size_t)9 * CO * CI;             // 294912
  if (ws_size < 2 * wt_elems * sizeof(bf16)) return;        // needs ~1.15 MB scratch
  bf16* wt_hi = (bf16*)d_ws;
  bf16* wt_lo = wt_hi + wt_elems;

  wt_transform<<<dim3((unsigned)(wt_elems / 256)), dim3(256), 0, stream>>>(wgt, wt_hi, wt_lo);

  dim3 grid(2, HH, NB);  // (co-block, h, b)
  conv_wmma<<<grid, dim3(256), 0, stream>>>(x, wt_hi, wt_lo, bias, out);
}
